// GraphSAGE_22617297781314
// MI455X (gfx1250) — matrix operand order repaired and checked by reading, not compiled, not run on hardware
//
#include <hip/hip_runtime.h>

typedef _Float16 v16h __attribute__((ext_vector_type(16)));
typedef _Float16 v8h  __attribute__((ext_vector_type(8)));
typedef float    v8f  __attribute__((ext_vector_type(8)));
typedef float    v4f  __attribute__((ext_vector_type(4)));

#define HIDDIM 256
#define NGRAPH 64

// ---------------- utility kernels ----------------

__global__ void sage_zero_f32(float* __restrict__ p, int n) {
  int i = blockIdx.x * blockDim.x + threadIdx.x;
  if (i < n) p[i] = 0.0f;
}

__global__ void sage_cvt_f32_f16(const float* __restrict__ in, _Float16* __restrict__ out, int n) {
  int i = blockIdx.x * blockDim.x + threadIdx.x;
  if (i < n) out[i] = (_Float16)in[i];
}

__global__ void sage_deg_kernel(const int* __restrict__ dst, float* __restrict__ deg, int nE) {
  int e = blockIdx.x * blockDim.x + threadIdx.x;
  if (e < nE) atomicAdd(deg + dst[e], 1.0f);
}

__global__ void sage_rdeg_kernel(const float* __restrict__ deg, float* __restrict__ rdeg, int n) {
  int i = blockIdx.x * blockDim.x + threadIdx.x;
  if (i < n) rdeg[i] = 1.0f / fmaxf(deg[i], 1.0f);
}

// ---------------- edge scatter (mean-agg numerator) ----------------
// one thread per (edge, 8-wide feature chunk); f16 gather, f32 atomic scatter
__global__ void sage_scatter_kernel(const _Float16* __restrict__ h,
                                    const int* __restrict__ src,
                                    const int* __restrict__ dst,
                                    float* __restrict__ agg,
                                    int nE, int din) {
  int idx = blockIdx.x * blockDim.x + threadIdx.x;
  int chunks = din >> 3;
  int e = idx / chunks;
  if (e >= nE) return;
  int c = (idx - e * chunks) << 3;
  int sn = src[e];
  int dn = dst[e];
  v8h v = *(const v8h*)(h + (size_t)sn * din + c);
  float* out = agg + (size_t)dn * din + c;
#pragma unroll
  for (int i = 0; i < 8; ++i) atomicAdd(out + i, (float)v[i]);
}

// ---------------- WMMA GEMM: hout = relu(mean(agg)@Wl^T + h@Wr^T + b), f16 out ---------
// grid (N/16), block 128 (4 waves). Each wave computes a 16x64 output strip:
// the A fragment (16 node rows) is loaded/converted ONCE per K-step and reused
// across 4 column tiles (4 WMMAs per K-step per pass).
__global__ __launch_bounds__(128)
void sage_gemm_kernel(const float* __restrict__ agg, const float* __restrict__ rdeg,
                      const _Float16* __restrict__ hin,
                      const _Float16* __restrict__ Wl, const _Float16* __restrict__ Wr,
                      const float* __restrict__ bias, _Float16* __restrict__ hout,
                      int N, int din) {
  const int dout = HIDDIM;
  int lane = threadIdx.x & 31;
  int wave = threadIdx.x >> 5;
  int mbase = blockIdx.x * 16;
  int nbase = wave * 64;               // 4 waves cover all 256 output columns
  int mr = lane & 15;
  int s  = lane >> 4;
  int arow = mbase + mr; if (arow >= N) arow = N - 1;

  float scale = rdeg[arow];
  v8f c0 = {}, c1 = {}, c2 = {}, c3 = {};

  // B row pointers: B[k][n] = W[n][k], W row-major [dout, din]; lane's column = nbase + t*16 + mr
  const _Float16* wl0 = Wl + (size_t)(nbase + 0 * 16 + mr) * din;
  const _Float16* wl1 = Wl + (size_t)(nbase + 1 * 16 + mr) * din;
  const _Float16* wl2 = Wl + (size_t)(nbase + 2 * 16 + mr) * din;
  const _Float16* wl3 = Wl + (size_t)(nbase + 3 * 16 + mr) * din;

  // ---- Pass 1: A = mean-scaled agg (f32 -> f16 on the fly), B = Wl ----
  const float* aggr = agg + (size_t)arow * din;
  for (int kt = 0; kt < din; kt += 32) {
    if (kt + 32 < din) __builtin_prefetch(aggr + kt + 32, 0, 3);  // global_prefetch_b8
    v4f a0 = *(const v4f*)(aggr + kt + s * 16);
    v4f a1 = *(const v4f*)(aggr + kt + s * 16 + 4);
    v4f a2 = *(const v4f*)(aggr + kt + s * 16 + 8);
    v4f a3 = *(const v4f*)(aggr + kt + s * 16 + 12);
    v16h a;
#pragma unroll
    for (int i = 0; i < 4; ++i) {
      a[i]      = (_Float16)(a0[i] * scale);
      a[4 + i]  = (_Float16)(a1[i] * scale);
      a[8 + i]  = (_Float16)(a2[i] * scale);
      a[12 + i] = (_Float16)(a3[i] * scale);
    }
    int ko = kt + s * 16;
    v16h b0 = *(const v16h*)(wl0 + ko);
    v16h b1 = *(const v16h*)(wl1 + ko);
    v16h b2 = *(const v16h*)(wl2 + ko);
    v16h b3 = *(const v16h*)(wl3 + ko);
    c0 = __builtin_amdgcn_wmma_f32_16x16x32_f16(false, a, false, b0, (short)0, c0, false, false);
    c1 = __builtin_amdgcn_wmma_f32_16x16x32_f16(false, a, false, b1, (short)0, c1, false, false);
    c2 = __builtin_amdgcn_wmma_f32_16x16x32_f16(false, a, false, b2, (short)0, c2, false, false);
    c3 = __builtin_amdgcn_wmma_f32_16x16x32_f16(false, a, false, b3, (short)0, c3, false, false);
  }

  // ---- Pass 2: A = hin (f16 direct), B = Wr ----
  const _Float16* hrow = hin + (size_t)arow * din;
  const _Float16* wr0 = Wr + (size_t)(nbase + 0 * 16 + mr) * din;
  const _Float16* wr1 = Wr + (size_t)(nbase + 1 * 16 + mr) * din;
  const _Float16* wr2 = Wr + (size_t)(nbase + 2 * 16 + mr) * din;
  const _Float16* wr3 = Wr + (size_t)(nbase + 3 * 16 + mr) * din;
  for (int kt = 0; kt < din; kt += 32) {
    v8h lo = *(const v8h*)(hrow + kt + s * 16);
    v8h hi = *(const v8h*)(hrow + kt + s * 16 + 8);
    v16h a;
#pragma unroll
    for (int i = 0; i < 8; ++i) { a[i] = lo[i]; a[8 + i] = hi[i]; }
    int ko = kt + s * 16;
    v16h b0 = *(const v16h*)(wr0 + ko);
    v16h b1 = *(const v16h*)(wr1 + ko);
    v16h b2 = *(const v16h*)(wr2 + ko);
    v16h b3 = *(const v16h*)(wr3 + ko);
    c0 = __builtin_amdgcn_wmma_f32_16x16x32_f16(false, a, false, b0, (short)0, c0, false, false);
    c1 = __builtin_amdgcn_wmma_f32_16x16x32_f16(false, a, false, b1, (short)0, c1, false, false);
    c2 = __builtin_amdgcn_wmma_f32_16x16x32_f16(false, a, false, b2, (short)0, c2, false, false);
    c3 = __builtin_amdgcn_wmma_f32_16x16x32_f16(false, a, false, b3, (short)0, c3, false, false);
  }

  // ---- epilogue: bias + relu + f16 store ----
  // C/D layout: lane l holds N = ntile + (l&15); VGPR r holds M = mbase + (l>>4)*8 + r
  v8f cc[4] = { c0, c1, c2, c3 };
#pragma unroll
  for (int t = 0; t < 4; ++t) {
    int nout = nbase + t * 16 + mr;
    float bv = bias[nout];
#pragma unroll
    for (int r = 0; r < 8; ++r) {
      int m = mbase + s * 8 + r;
      if (m < N) {
        float v = cc[t][r] + bv;
        v = v > 0.0f ? v : 0.0f;
        hout[(size_t)m * dout + nout] = (_Float16)v;
      }
    }
  }
}

// ---------------- in-place row L2 normalize (f16), one 256-thread block per row ------
__global__ __launch_bounds__(256)
void sage_l2norm_kernel(_Float16* __restrict__ h, int N) {
  __shared__ float red[256];
  int row = blockIdx.x;
  if (row >= N) return;
  float v = (float)h[(size_t)row * HIDDIM + threadIdx.x];
  red[threadIdx.x] = v * v;
  __syncthreads();
  for (int off = 128; off > 0; off >>= 1) {
    if (threadIdx.x < off) red[threadIdx.x] += red[threadIdx.x + off];
    __syncthreads();
  }
  float inv = 1.0f / fmaxf(sqrtf(red[0]), 1e-12f);
  h[(size_t)row * HIDDIM + threadIdx.x] = (_Float16)(v * inv);
}

// ---------------- global mean pool ----------------
__global__ void sage_pool_kernel(const _Float16* __restrict__ h, const int* __restrict__ batch,
                                 float* __restrict__ gsum, float* __restrict__ cnt, int N) {
  int idx = blockIdx.x * blockDim.x + threadIdx.x;
  int n = idx >> 5;           // 32 chunks of 8 per node
  if (n >= N) return;
  int c = (idx & 31) << 3;
  int g = batch[n];
  v8h v = *(const v8h*)(h + (size_t)n * HIDDIM + c);
  float* out = gsum + (size_t)g * HIDDIM + c;
#pragma unroll
  for (int i = 0; i < 8; ++i) atomicAdd(out + i, (float)v[i]);
  if ((idx & 31) == 0) atomicAdd(cnt + g, 1.0f);
}

// ---------------- classifier ----------------
__global__ __launch_bounds__(128)
void sage_cls1_kernel(const float* __restrict__ gsum, const float* __restrict__ cnt,
                      const float* __restrict__ cW1, const float* __restrict__ cb1,
                      float* __restrict__ hcls) {
  int g = blockIdx.x;           // 64 graphs
  int j = threadIdx.x;          // 128 hidden
  float inv = 1.0f / fmaxf(cnt[g], 1.0f);
  const float* grow = gsum + (size_t)g * HIDDIM;
  const float* wrow = cW1 + (size_t)j * HIDDIM;
  float acc = 0.0f;
  for (int k = 0; k < HIDDIM; ++k) acc += grow[k] * inv * wrow[k];
  acc += cb1[j];
  hcls[(size_t)g * 128 + j] = acc > 0.0f ? acc : 0.0f;
}

__global__ __launch_bounds__(128)
void sage_cls2_kernel(const float* __restrict__ hcls, const float* __restrict__ cW2,
                      const float* __restrict__ cb2, float* __restrict__ out) {
  int t = threadIdx.x;          // 64 graphs x 2 outputs
  if (t >= NGRAPH * 2) return;
  int g = t >> 1, o = t & 1;
  const float* hrow = hcls + (size_t)g * 128;
  const float* wrow = cW2 + (size_t)o * 128;
  float acc = 0.0f;
  for (int k = 0; k < 128; ++k) acc += hrow[k] * wrow[k];
  out[g * 2 + o] = acc + cb2[o];
}

// ---------------- host launch ----------------

extern "C" void kernel_launch(void* const* d_in, const int* in_sizes, int n_in,
                              void* d_out, int out_size, void* d_ws, size_t ws_size,
                              hipStream_t stream) {
  const int IN_DIM = 32;
  const float* x    = (const float*)d_in[0];
  const int* ei     = (const int*)d_in[1];
  const int* batch  = (const int*)d_in[2];
  const float* Wl[3] = { (const float*)d_in[3], (const float*)d_in[6], (const float*)d_in[9]  };
  const float* Wr[3] = { (const float*)d_in[4], (const float*)d_in[7], (const float*)d_in[10] };
  const float* bb[3] = { (const float*)d_in[5], (const float*)d_in[8], (const float*)d_in[11] };
  const float* cW1 = (const float*)d_in[12];
  const float* cb1 = (const float*)d_in[13];
  const float* cW2 = (const float*)d_in[14];
  const float* cb2 = (const float*)d_in[15];
  float* out = (float*)d_out;

  const int N  = in_sizes[0] / IN_DIM;
  const int nE = in_sizes[1] / 2;
  const int* src = ei;
  const int* dst = ei + nE;

  // workspace partition (256B aligned)
  char* base = (char*)d_ws;
  size_t off = 0;
  auto take = [&](size_t bytes) { char* p = base + off; off = (off + bytes + 255) & ~(size_t)255; return p; };
  float*    agg  = (float*)   take((size_t)N * HIDDIM * 4);
  _Float16* h_a  = (_Float16*)take((size_t)N * HIDDIM * 2);
  _Float16* h_b  = (_Float16*)take((size_t)N * HIDDIM * 2);
  float*    deg  = (float*)   take((size_t)N * 4);
  float*    rdeg = (float*)   take((size_t)N * 4);
  _Float16* wl16[3]; _Float16* wr16[3];
  const int dins[3] = { IN_DIM, HIDDIM, HIDDIM };
  for (int l = 0; l < 3; ++l) {
    wl16[l] = (_Float16*)take((size_t)HIDDIM * dins[l] * 2);
    wr16[l] = (_Float16*)take((size_t)HIDDIM * dins[l] * 2);
  }
  float* gsum = (float*)take((size_t)NGRAPH * HIDDIM * 4);
  float* cnt  = (float*)take((size_t)NGRAPH * 4);
  float* hcls = (float*)take((size_t)NGRAPH * 128 * 4);
  (void)ws_size; (void)n_in; (void)out_size;

  const int TB = 256;
  auto blocks = [](long long n, int tb) { return (unsigned)((n + tb - 1) / tb); };

  // degree (once)
  sage_zero_f32<<<blocks(N, TB), TB, 0, stream>>>(deg, N);
  sage_deg_kernel<<<blocks(nE, TB), TB, 0, stream>>>(dst, deg, nE);
  sage_rdeg_kernel<<<blocks(N, TB), TB, 0, stream>>>(deg, rdeg, N);

  // convert input features and weights to f16
  sage_cvt_f32_f16<<<blocks((long long)N * IN_DIM, TB), TB, 0, stream>>>(x, h_a, N * IN_DIM);
  for (int l = 0; l < 3; ++l) {
    int wn = HIDDIM * dins[l];
    sage_cvt_f32_f16<<<blocks(wn, TB), TB, 0, stream>>>(Wl[l], wl16[l], wn);
    sage_cvt_f32_f16<<<blocks(wn, TB), TB, 0, stream>>>(Wr[l], wr16[l], wn);
  }

  // 3 SAGE layers, ping-pong h_a / h_b
  _Float16* hcur = h_a;
  _Float16* hnxt = h_b;
  for (int l = 0; l < 3; ++l) {
    int din = dins[l];
    sage_zero_f32<<<blocks((long long)N * din, TB), TB, 0, stream>>>(agg, N * din);
    long long sthreads = (long long)nE * (din >> 3);
    sage_scatter_kernel<<<blocks(sthreads, TB), TB, 0, stream>>>(hcur, src, dst, agg, nE, din);
    sage_gemm_kernel<<<(N + 15) / 16, 128, 0, stream>>>(agg, rdeg, hcur, wl16[l], wr16[l], bb[l], hnxt, N, din);
    sage_l2norm_kernel<<<N, 256, 0, stream>>>(hnxt, N);
    _Float16* t = hcur; hcur = hnxt; hnxt = t;
  }
  // hcur now holds final node embeddings [N, 256] f16

  // pooling + classifier
  sage_zero_f32<<<blocks(NGRAPH * HIDDIM, TB), TB, 0, stream>>>(gsum, NGRAPH * HIDDIM);
  sage_zero_f32<<<1, NGRAPH, 0, stream>>>(cnt, NGRAPH);
  sage_pool_kernel<<<blocks((long long)N * 32, TB), TB, 0, stream>>>(hcur, batch, gsum, cnt, N);
  sage_cls1_kernel<<<NGRAPH, 128, 0, stream>>>(gsum, cnt, cW1, cb1, hcls);
  sage_cls2_kernel<<<1, 128, 0, stream>>>(hcls, cW2, cb2, out);
}
